// Model_47510928229119
// MI455X (gfx1250) — compile-verified, matrix operand order checked by reading, not run on hardware
//
#include <hip/hip_runtime.h>
#include <hip/hip_bf16.h>

#define Bb 32
#define Ss 128
#define Uu 8192
#define Ll 16
#define WEe 256
#define CEe 64
#define Hh 256
#define G3 768
#define Tt 1024

typedef __attribute__((ext_vector_type(16))) __bf16 v16bf;
typedef __attribute__((ext_vector_type(8)))  float  v8f;

__device__ __forceinline__ v8f wmma_bf16(v16bf a, v16bf b, v8f c) {
  return __builtin_amdgcn_wmma_f32_16x16x32_bf16(false, a, false, b, (short)0, c, false, false);
}

// Load a 16x32 bf16 fragment (A from row-major [16][ldk], or B from Bt=[N][K] row-major).
// Lane l covers row (row0 + l%16); halves 0-7 = K kbase..kbase+7, halves 8-15 = kbase+16..kbase+23,
// kbase = k0 + (l/16)*8.  All addresses 16B aligned (ldk multiple of 8, k0 multiple of 32).
__device__ __forceinline__ v16bf load_frag(const __bf16* base, int row0, int ldk, int k0, int lane) {
  int r  = row0 + (lane & 15);
  int kb = k0 + ((lane >> 4) << 3);
  const __bf16* p = base + (size_t)r * ldk + kb;
  v16bf f;
  ((uint4*)&f)[0] = *(const uint4*)(p);
  ((uint4*)&f)[1] = *(const uint4*)(p + 16);
  return f;
}

__device__ __forceinline__ float sigmoidf_(float x) { return 1.0f / (1.0f + __expf(-x)); }

// ---------------- prep ----------------
__global__ void k_transpose_bf16(const float* __restrict__ src, __bf16* __restrict__ dst, int K, int N) {
  int idx = blockIdx.x * blockDim.x + threadIdx.x;
  if (idx >= K * N) return;
  int k = idx / N, n = idx % N;
  dst[(size_t)n * K + k] = (__bf16)src[idx];
}

__global__ void k_zero(unsigned int* p, int n) {
  int i = blockIdx.x * blockDim.x + threadIdx.x;
  if (i < n) p[i] = 0u;
}

// ---------------- char embedding gather (fwd + ragged-reversed) ----------------
__global__ void k_char_gather(const int* __restrict__ char_ids, const int* __restrict__ char_len,
                              const float* __restrict__ char_emb,
                              __bf16* __restrict__ cef, __bf16* __restrict__ ceb) {
  int idx = blockIdx.x * blockDim.x + threadIdx.x;
  if (idx >= Uu * Ll * CEe) return;
  int e = idx % CEe;
  int t = (idx / CEe) % Ll;
  int u = idx / (CEe * Ll);
  int c = char_ids[u * Ll + t];
  __bf16 v = (__bf16)char_emb[(size_t)c * CEe + e];
  cef[idx] = v;
  int len = char_len[u];
  int tr  = (t < len) ? (len - 1 - t) : t;
  ceb[((size_t)u * Ll + tr) * CEe + e] = v;
}

// ---------------- fused char GRU step: rec GEMM + input-proj GEMM + gate ----------------
__global__ __launch_bounds__(512) void k_char_step(
    const __bf16* __restrict__ cef, const __bf16* __restrict__ ceb,
    const __bf16* __restrict__ UwTf, const __bf16* __restrict__ UwTb,
    const __bf16* __restrict__ WTf,  const __bf16* __restrict__ WTb,
    const float* __restrict__ cbf_, const float* __restrict__ cbb_,
    const int* __restrict__ char_len,
    float* __restrict__ hf32f, float* __restrict__ hf32b,
    __bf16* __restrict__ hb16f, __bf16* __restrict__ hb16b, int t) {
  const int dir = blockIdx.y;
  const __bf16* ce   = dir ? ceb  : cef;
  const __bf16* UwT  = dir ? UwTb : UwTf;
  const __bf16* WT   = dir ? WTb  : WTf;
  const float*  bias = dir ? cbb_ : cbf_;
  float*  hf32 = dir ? hf32b : hf32f;
  __bf16* hb16 = dir ? hb16b : hb16f;

  __shared__ __align__(16) __bf16 hA[16 * 256];
  __shared__ __align__(16) __bf16 xA[16 * 64];
  __shared__ int lenS[16];

  const int tid  = threadIdx.x;
  const int row0 = blockIdx.x * 16;

  // 16x256 bf16 = 512 uint4 (one per thread)
  ((uint4*)hA)[tid] = ((const uint4*)(hb16 + (size_t)row0 * 256))[tid];
  if (tid < 128) {
    int u = tid >> 3, part = tid & 7;
    const uint4* s = (const uint4*)(ce + (((size_t)(row0 + u)) * Ll + t) * CEe);
    ((uint4*)xA)[tid] = s[part];
  }
  if (tid < 16) lenS[tid] = char_len[row0 + tid];
  __syncthreads();

  const int wv = tid >> 5, lane = tid & 31;
  const int j0 = wv * 16, j1 = 256 + j0, j2 = 512 + j0;
  v8f az = {}, ar = {}, arh = {}, axh = {};

  for (int kb = 0; kb < 256; kb += 32) {
    v16bf a = load_frag(hA, 0, 256, kb, lane);
    az  = wmma_bf16(a, load_frag(UwT, j0, 256, kb, lane), az);
    ar  = wmma_bf16(a, load_frag(UwT, j1, 256, kb, lane), ar);
    arh = wmma_bf16(a, load_frag(UwT, j2, 256, kb, lane), arh);
  }
  for (int kb = 0; kb < 64; kb += 32) {
    v16bf a = load_frag(xA, 0, 64, kb, lane);
    az  = wmma_bf16(a, load_frag(WT, j0, 64, kb, lane), az);
    ar  = wmma_bf16(a, load_frag(WT, j1, 64, kb, lane), ar);
    axh = wmma_bf16(a, load_frag(WT, j2, 64, kb, lane), axh);
  }

  const int n  = lane & 15;
  const int mb = (lane >> 4) << 3;
  const float bz = bias[j0 + n] + bias[G3 + j0 + n];
  const float br = bias[j1 + n] + bias[G3 + j1 + n];
  const float b0h = bias[j2 + n], b1h = bias[G3 + j2 + n];
#pragma unroll
  for (int v = 0; v < 8; v++) {
    int m = v + mb;
    size_t hidx = (size_t)(row0 + m) * 256 + j0 + n;
    float hold = hf32[hidx];
    float z  = sigmoidf_(az[v] + bz);
    float r  = sigmoidf_(ar[v] + br);
    float hh = tanhf(axh[v] + b0h + r * (arh[v] + b1h));
    float hn = (t < lenS[m]) ? (z * hold + (1.0f - z) * hh) : hold;
    hf32[hidx] = hn;
    hb16[hidx] = (__bf16)hn;
  }
}

// ---------------- build word-GRU input x = [word_emb | hf | hb] (fwd + reversed) ----------------
__global__ void k_word_x(const int* __restrict__ word_ids, const int* __restrict__ uniq_idx,
                         const int* __restrict__ seq_len, const float* __restrict__ word_emb,
                         const __bf16* __restrict__ hcf, const __bf16* __restrict__ hcb,
                         __bf16* __restrict__ x, __bf16* __restrict__ xr) {
  int idx = blockIdx.x * blockDim.x + threadIdx.x;
  if (idx >= Bb * Ss * G3) return;
  int e = idx % G3;
  int s = (idx / G3) % Ss;
  int b = idx / (G3 * Ss);
  __bf16 v;
  if (e < 256)      v = (__bf16)word_emb[(size_t)word_ids[b * Ss + s] * WEe + e];
  else if (e < 512) v = hcf[(size_t)uniq_idx[b * Ss + s] * 256 + (e - 256)];
  else              v = hcb[(size_t)uniq_idx[b * Ss + s] * 256 + (e - 512)];
  x[idx] = v;
  int len = seq_len[b];
  int sr  = (s < len) ? (len - 1 - s) : s;
  xr[((size_t)b * Ss + sr) * G3 + e] = v;
}

// ---------------- word input projection: xz = x @ wW + b[0] ----------------
__global__ __launch_bounds__(512) void k_word_proj(
    const __bf16* __restrict__ xf, const __bf16* __restrict__ xr,
    const __bf16* __restrict__ WTf, const __bf16* __restrict__ WTb,
    const float* __restrict__ bf_, const float* __restrict__ bb_,
    float* __restrict__ xzf, float* __restrict__ xzb) {
  const int dir = blockIdx.y;
  const __bf16* x    = dir ? xr   : xf;
  const __bf16* WT   = dir ? WTb  : WTf;
  const float*  bias = dir ? bb_  : bf_;
  float* out = dir ? xzb : xzf;

  __shared__ __align__(16) __bf16 aS[16 * G3];
  const int tid = threadIdx.x;
  const int row0 = blockIdx.x * 16;
  const uint4* src = (const uint4*)(x + (size_t)row0 * G3);
  for (int i = tid; i < 1536; i += 512) ((uint4*)aS)[i] = src[i];
  __syncthreads();

  const int wv = tid >> 5, lane = tid & 31;
#pragma unroll
  for (int i = 0; i < 3; i++) {
    int n0 = wv * 16 + i * 256;
    v8f acc = {};
    for (int kb = 0; kb < G3; kb += 32) {
      v16bf a = load_frag(aS, 0, G3, kb, lane);
      acc = wmma_bf16(a, load_frag(WT, n0, G3, kb, lane), acc);
    }
    int n = lane & 15, mb = (lane >> 4) << 3;
    float bv = bias[n0 + n];
#pragma unroll
    for (int v = 0; v < 8; v++)
      out[((size_t)(row0 + v + mb)) * G3 + n0 + n] = acc[v] + bv;
  }
}

// ---------------- fused word GRU step (rec GEMM + gate, xz precomputed) ----------------
__global__ __launch_bounds__(512) void k_word_step(
    const float* __restrict__ xzf, const float* __restrict__ xzb,
    const __bf16* __restrict__ UwTf, const __bf16* __restrict__ UwTb,
    const float* __restrict__ bf_, const float* __restrict__ bb_,
    const int* __restrict__ seq_len,
    float* __restrict__ hwf, float* __restrict__ hwb,
    __bf16* __restrict__ hbf, __bf16* __restrict__ hbb,
    float* __restrict__ yf, float* __restrict__ yb, int t) {
  const int dir = blockIdx.y;
  const float*  xz   = dir ? xzb  : xzf;
  const __bf16* UwT  = dir ? UwTb : UwTf;
  const float*  bias = dir ? bb_  : bf_;
  float*  hf32 = dir ? hwb : hwf;
  __bf16* hb16 = dir ? hbb : hbf;
  float*  y    = dir ? yb  : yf;

  __shared__ __align__(16) __bf16 hA[16 * 256];
  __shared__ int lenS[16];
  const int tid = threadIdx.x;
  const int row0 = blockIdx.x * 16;
  ((uint4*)hA)[tid] = ((const uint4*)(hb16 + (size_t)row0 * 256))[tid];
  if (tid < 16) lenS[tid] = seq_len[row0 + tid];
  __syncthreads();

  const int wv = tid >> 5, lane = tid & 31;
  const int j0 = wv * 16, j1 = 256 + j0, j2 = 512 + j0;
  v8f az = {}, ar = {}, arh = {};
  for (int kb = 0; kb < 256; kb += 32) {
    v16bf a = load_frag(hA, 0, 256, kb, lane);
    az  = wmma_bf16(a, load_frag(UwT, j0, 256, kb, lane), az);
    ar  = wmma_bf16(a, load_frag(UwT, j1, 256, kb, lane), ar);
    arh = wmma_bf16(a, load_frag(UwT, j2, 256, kb, lane), arh);
  }
  const int n = lane & 15, mb = (lane >> 4) << 3;
  const float b1z = bias[G3 + j0 + n];
  const float b1r = bias[G3 + j1 + n];
  const float b1h = bias[G3 + j2 + n];
#pragma unroll
  for (int v = 0; v < 8; v++) {
    int m = v + mb;
    int b = row0 + m;
    const float* xzrow = xz + ((size_t)b * Ss + t) * G3;
    size_t hidx = (size_t)b * 256 + j0 + n;
    float hold = hf32[hidx];
    float z  = sigmoidf_(xzrow[j0 + n] + az[v] + b1z);
    float r  = sigmoidf_(xzrow[j1 + n] + ar[v] + b1r);
    float hh = tanhf(xzrow[j2 + n] + r * (arh[v] + b1h));
    int len = lenS[m];
    bool mt = t < len;
    float hn = mt ? (z * hold + (1.0f - z) * hh) : hold;
    hf32[hidx] = hn;
    hb16[hidx] = (__bf16)hn;
    int outpos = dir ? (mt ? (len - 1 - t) : t) : t;  // bwd writes into reversed-valid slot
    y[((size_t)b * Ss + outpos) * 256 + j0 + n] = mt ? hn : 0.0f;
  }
}

// ---------------- dense: logits = (yf+yb) @ dW + db ----------------
__global__ __launch_bounds__(512) void k_dense(
    const float* __restrict__ yf, const float* __restrict__ yb,
    const __bf16* __restrict__ dWT, const float* __restrict__ db,
    float* __restrict__ logits) {
  __shared__ __align__(16) __bf16 aS[16 * 256];
  const int tid = threadIdx.x;
  const int row0 = blockIdx.x * 16;
  for (int i = tid; i < 16 * 256; i += 512)
    aS[i] = (__bf16)(yf[(size_t)row0 * 256 + i] + yb[(size_t)row0 * 256 + i]);
  __syncthreads();
  const int wv = tid >> 5, lane = tid & 31;
#pragma unroll
  for (int i = 0; i < 4; i++) {
    int n0 = wv * 64 + i * 16;
    v8f acc = {};
    for (int kb = 0; kb < 256; kb += 32) {
      v16bf a = load_frag(aS, 0, 256, kb, lane);
      acc = wmma_bf16(a, load_frag(dWT, n0, 256, kb, lane), acc);
    }
    int n = lane & 15, mb = (lane >> 4) << 3;
    float bv = db[n0 + n];
#pragma unroll
    for (int v = 0; v < 8; v++)
      logits[((size_t)(row0 + v + mb)) * Tt + n0 + n] = acc[v] + bv;
  }
}

// ---------------- softmax (wave per row) * wmask ----------------
__global__ __launch_bounds__(256) void k_softmax(const float* __restrict__ logits,
                                                 const int* __restrict__ seq_len,
                                                 float* __restrict__ out) {
  const int tid = threadIdx.x;
  const int wv = tid >> 5, lane = tid & 31;
  const int row = blockIdx.x * 8 + wv;
  const int b = row / Ss, s = row % Ss;
  const bool valid = s < seq_len[b];
  const float* lr = logits + (size_t)row * Tt;
  float vals[32];
  float mx = -1e30f;
#pragma unroll
  for (int i = 0; i < 32; i++) { vals[i] = lr[lane + i * 32]; mx = fmaxf(mx, vals[i]); }
  for (int o = 16; o > 0; o >>= 1) mx = fmaxf(mx, __shfl_xor(mx, o, 32));
  float sum = 0.0f;
#pragma unroll
  for (int i = 0; i < 32; i++) { vals[i] = __expf(vals[i] - mx); sum += vals[i]; }
  for (int o = 16; o > 0; o >>= 1) sum += __shfl_xor(sum, o, 32);
  float inv = valid ? (1.0f / sum) : 0.0f;
  float* orow = out + (size_t)row * Tt;
#pragma unroll
  for (int i = 0; i < 32; i++) orow[lane + i * 32] = vals[i] * inv;
}

extern "C" void kernel_launch(void* const* d_in, const int* in_sizes, int n_in,
                              void* d_out, int out_size, void* d_ws, size_t ws_size,
                              hipStream_t stream) {
  const int*   word_ids = (const int*)d_in[0];
  const int*   uniq_idx = (const int*)d_in[1];
  const int*   char_ids = (const int*)d_in[2];
  const int*   seq_len  = (const int*)d_in[3];
  const int*   char_len = (const int*)d_in[4];
  const float* word_emb = (const float*)d_in[5];
  const float* char_emb = (const float*)d_in[6];
  const float* cWf = (const float*)d_in[7];
  const float* cUf = (const float*)d_in[8];
  const float* cbf = (const float*)d_in[9];
  const float* cWb = (const float*)d_in[10];
  const float* cUb = (const float*)d_in[11];
  const float* cbb = (const float*)d_in[12];
  const float* wWf = (const float*)d_in[13];
  const float* wUf = (const float*)d_in[14];
  const float* wbf = (const float*)d_in[15];
  const float* wWb = (const float*)d_in[16];
  const float* wUb = (const float*)d_in[17];
  const float* wbb = (const float*)d_in[18];
  const float* dW  = (const float*)d_in[19];
  const float* db  = (const float*)d_in[20];
  (void)in_sizes; (void)n_in; (void)out_size; (void)ws_size;

  char* base = (char*)d_ws;
  size_t off = 0;
  auto alloc = [&](size_t bytes) -> void* {
    off = (off + 255) & ~(size_t)255;
    void* p = base + off;
    off += bytes;
    return p;
  };

  __bf16* cUfT = (__bf16*)alloc((size_t)G3 * 256 * 2);
  __bf16* cUbT = (__bf16*)alloc((size_t)G3 * 256 * 2);
  __bf16* wUfT = (__bf16*)alloc((size_t)G3 * 256 * 2);
  __bf16* wUbT = (__bf16*)alloc((size_t)G3 * 256 * 2);
  __bf16* cWfT = (__bf16*)alloc((size_t)G3 * 64 * 2);
  __bf16* cWbT = (__bf16*)alloc((size_t)G3 * 64 * 2);
  __bf16* wWfT = (__bf16*)alloc((size_t)G3 * G3 * 2);
  __bf16* wWbT = (__bf16*)alloc((size_t)G3 * G3 * 2);
  __bf16* dWT  = (__bf16*)alloc((size_t)Tt * 256 * 2);
  __bf16* ce_f = (__bf16*)alloc((size_t)Uu * Ll * CEe * 2);
  __bf16* ce_b = (__bf16*)alloc((size_t)Uu * Ll * CEe * 2);
  __bf16* hcb16_f = (__bf16*)alloc((size_t)Uu * 256 * 2);
  __bf16* hcb16_b = (__bf16*)alloc((size_t)Uu * 256 * 2);
  float*  hc_f = (float*)alloc((size_t)Uu * 256 * 4);
  float*  hc_b = (float*)alloc((size_t)Uu * 256 * 4);
  __bf16* x_f = (__bf16*)alloc((size_t)Bb * Ss * G3 * 2);
  __bf16* x_b = (__bf16*)alloc((size_t)Bb * Ss * G3 * 2);
  float*  xz_f = (float*)alloc((size_t)Bb * Ss * G3 * 4);
  float*  xz_b = (float*)alloc((size_t)Bb * Ss * G3 * 4);
  float*  hw_f = (float*)alloc((size_t)Bb * 256 * 4);
  float*  hw_b = (float*)alloc((size_t)Bb * 256 * 4);
  __bf16* hwb16_f = (__bf16*)alloc((size_t)Bb * 256 * 2);
  __bf16* hwb16_b = (__bf16*)alloc((size_t)Bb * 256 * 2);
  float*  yf = (float*)alloc((size_t)Bb * Ss * 256 * 4);
  float*  yb = (float*)alloc((size_t)Bb * Ss * 256 * 4);
  float*  logits = (float*)alloc((size_t)Bb * Ss * Tt * 4);

  const int TPB = 256;
  auto cdiv = [](int a, int b) { return (a + b - 1) / b; };

  // weight prep (transpose + bf16)
  k_transpose_bf16<<<cdiv(256 * G3, TPB), TPB, 0, stream>>>(cUf, cUfT, 256, G3);
  k_transpose_bf16<<<cdiv(256 * G3, TPB), TPB, 0, stream>>>(cUb, cUbT, 256, G3);
  k_transpose_bf16<<<cdiv(256 * G3, TPB), TPB, 0, stream>>>(wUf, wUfT, 256, G3);
  k_transpose_bf16<<<cdiv(256 * G3, TPB), TPB, 0, stream>>>(wUb, wUbT, 256, G3);
  k_transpose_bf16<<<cdiv(64 * G3, TPB), TPB, 0, stream>>>(cWf, cWfT, 64, G3);
  k_transpose_bf16<<<cdiv(64 * G3, TPB), TPB, 0, stream>>>(cWb, cWbT, 64, G3);
  k_transpose_bf16<<<cdiv(G3 * G3, TPB), TPB, 0, stream>>>(wWf, wWfT, G3, G3);
  k_transpose_bf16<<<cdiv(G3 * G3, TPB), TPB, 0, stream>>>(wWb, wWbT, G3, G3);
  k_transpose_bf16<<<cdiv(256 * Tt, TPB), TPB, 0, stream>>>(dW, dWT, 256, Tt);

  // zero initial hidden states (f32 masters + bf16 copies)
  k_zero<<<cdiv(Uu * 256, TPB), TPB, 0, stream>>>((unsigned int*)hc_f, Uu * 256);
  k_zero<<<cdiv(Uu * 256, TPB), TPB, 0, stream>>>((unsigned int*)hc_b, Uu * 256);
  k_zero<<<cdiv(Uu * 128, TPB), TPB, 0, stream>>>((unsigned int*)hcb16_f, Uu * 128);
  k_zero<<<cdiv(Uu * 128, TPB), TPB, 0, stream>>>((unsigned int*)hcb16_b, Uu * 128);
  k_zero<<<cdiv(Bb * 256, TPB), TPB, 0, stream>>>((unsigned int*)hw_f, Bb * 256);
  k_zero<<<cdiv(Bb * 256, TPB), TPB, 0, stream>>>((unsigned int*)hw_b, Bb * 256);
  k_zero<<<cdiv(Bb * 128, TPB), TPB, 0, stream>>>((unsigned int*)hwb16_f, Bb * 128);
  k_zero<<<cdiv(Bb * 128, TPB), TPB, 0, stream>>>((unsigned int*)hwb16_b, Bb * 128);

  // char embeddings (fwd + ragged reverse)
  k_char_gather<<<cdiv(Uu * Ll * CEe, TPB), TPB, 0, stream>>>(char_ids, char_len, char_emb, ce_f, ce_b);

  // char BiGRU recurrence: 16 fused WMMA steps, both directions per launch
  for (int t = 0; t < Ll; t++) {
    k_char_step<<<dim3(Uu / 16, 2), 512, 0, stream>>>(
        ce_f, ce_b, cUfT, cUbT, cWfT, cWbT, cbf, cbb, char_len,
        hc_f, hc_b, hcb16_f, hcb16_b, t);
  }

  // word input x = [we | hf | hb] (fwd + ragged reverse)
  k_word_x<<<cdiv(Bb * Ss * G3, TPB), TPB, 0, stream>>>(
      word_ids, uniq_idx, seq_len, word_emb, hcb16_f, hcb16_b, x_f, x_b);

  // word input projection
  k_word_proj<<<dim3(Bb * Ss / 16, 2), 512, 0, stream>>>(
      x_f, x_b, wWfT, wWbT, wbf, wbb, xz_f, xz_b);

  // word BiGRU recurrence: 128 fused WMMA steps, both directions per launch
  for (int t = 0; t < Ss; t++) {
    k_word_step<<<dim3(Bb / 16, 2), 512, 0, stream>>>(
        xz_f, xz_b, wUfT, wUbT, wbf, wbb, seq_len,
        hw_f, hw_b, hwb16_f, hwb16_b, yf, yb, t);
  }

  // dense + softmax*mask
  k_dense<<<Bb * Ss / 16, 512, 0, stream>>>(yf, yb, dWT, db, logits);
  k_softmax<<<Bb * Ss / 8, 256, 0, stream>>>(logits, seq_len, (float*)d_out);
}
